// Pix2StructTextAttention_55336358643022
// MI455X (gfx1250) — matrix operand order repaired and checked by reading, not compiled, not run on hardware
//
#include <hip/hip_runtime.h>
#include <hip/hip_bf16.h>

// ---------------------------------------------------------------------------
// Types for CDNA5 WMMA (wave32, v_wmma_f32_16x16x32_bf16)
// ---------------------------------------------------------------------------
typedef __attribute__((ext_vector_type(16))) __bf16 v16bf;
typedef __attribute__((ext_vector_type(8)))  __bf16 bf16x8;
typedef __attribute__((ext_vector_type(4)))  __bf16 bf16x4;
typedef __attribute__((ext_vector_type(8)))  float  v8f;

__device__ __forceinline__ v8f wmma_bf16(v16bf a, v16bf b, v8f c) {
    // (neg_a, A, neg_b, B, c_mod, C, reuse_a, reuse_b)
    return __builtin_amdgcn_wmma_f32_16x16x32_bf16(false, a, false, b,
                                                   (short)0, c, false, false);
}

// ---------------------------------------------------------------------------
// gfx1250 async global->LDS copy (ASYNCcnt path), with graceful fallback
// ---------------------------------------------------------------------------
#if defined(__AMDGCN__) && __has_builtin(__builtin_amdgcn_global_load_async_to_lds_b128) && \
    __has_builtin(__builtin_amdgcn_s_wait_asynccnt)
#define USE_ASYNC_COPY 1
#else
#define USE_ASYNC_COPY 0
#endif

#if USE_ASYNC_COPY
// Builtin signature (from clang diagnostic): params are pointers to
// 'int __attribute__((vector_size(16)))' in global (AS1) / LDS (AS3) space.
typedef int v4i_vs __attribute__((vector_size(4 * sizeof(int))));
typedef __attribute__((address_space(1))) v4i_vs* gv4i_p;
typedef __attribute__((address_space(3))) v4i_vs* lv4i_p;
#endif

// Copy 64 contiguous bytes (4 x b128) from global to LDS for this lane.
__device__ __forceinline__ void stage_cp64(const __bf16* g, __bf16* l) {
#if USE_ASYNC_COPY
    gv4i_p gp = (gv4i_p)g;
    lv4i_p lp = (lv4i_p)l;
    // INST_OFFSET is added to BOTH the global and LDS address (ISA 10.x async ops)
    __builtin_amdgcn_global_load_async_to_lds_b128(gp, lp, 0,  0);
    __builtin_amdgcn_global_load_async_to_lds_b128(gp, lp, 16, 0);
    __builtin_amdgcn_global_load_async_to_lds_b128(gp, lp, 32, 0);
    __builtin_amdgcn_global_load_async_to_lds_b128(gp, lp, 48, 0);
#else
#pragma unroll
    for (int i = 0; i < 4; ++i)
        *(bf16x8*)(l + i * 8) = *(const bf16x8*)(g + i * 8);
#endif
}

__device__ __forceinline__ void stage_wait() {
#if USE_ASYNC_COPY
    __builtin_amdgcn_s_wait_asynccnt(0);
#endif
}

// ---------------------------------------------------------------------------
// Problem constants
// ---------------------------------------------------------------------------
#define BB    2
#define SQ    2048
#define HIDN  2048
#define NH    32
#define DKV   64
#define MROWS (BB * SQ)      // 4096
#define NQKV  (3 * HIDN)     // 6144

// Workspace layout (bytes)
static constexpr size_t HB_OFF    = 0;                               // bf16 [4096][2048]
static constexpr size_t HB_SZ     = (size_t)MROWS * HIDN * 2;
static constexpr size_t WQKVT_OFF = HB_OFF + HB_SZ;                  // bf16 [6144][2048] (transposed)
static constexpr size_t WQKVT_SZ  = (size_t)NQKV * HIDN * 2;
static constexpr size_t WOT_OFF   = WQKVT_OFF + WQKVT_SZ;            // bf16 [2048][2048] (transposed)
static constexpr size_t WOT_SZ    = (size_t)HIDN * HIDN * 2;
static constexpr size_t QB_OFF    = WOT_OFF + WOT_SZ;                // bf16 [b][h][s][d]
static constexpr size_t QB_SZ     = (size_t)BB * NH * SQ * DKV * 2;
static constexpr size_t KB_OFF    = QB_OFF + QB_SZ;                  // bf16 [b][h][s][d]
static constexpr size_t KB_SZ     = QB_SZ;
static constexpr size_t VT_OFF    = KB_OFF + KB_SZ;                  // bf16 [b][h][d][s]
static constexpr size_t VT_SZ     = QB_SZ;
static constexpr size_t AB_OFF    = VT_OFF + VT_SZ;                  // bf16 [4096][2048] attn out
static constexpr size_t AB_SZ     = (size_t)MROWS * HIDN * 2;
static constexpr size_t BIAS_OFF  = AB_OFF + AB_SZ;                  // f32  [32][2048]
// total ~117.7 MB

// ---------------------------------------------------------------------------
// Prep kernels
// ---------------------------------------------------------------------------
__global__ __launch_bounds__(256) void cvt_h_kernel(const float* __restrict__ H,
                                                    __bf16* __restrict__ Hb) {
    size_t i = ((size_t)blockIdx.x * blockDim.x + threadIdx.x) * 4;
    float4 v = *(const float4*)(H + i);
    bf16x4 o;
    o[0] = (__bf16)v.x; o[1] = (__bf16)v.y; o[2] = (__bf16)v.z; o[3] = (__bf16)v.w;
    *(bf16x4*)(Hb + i) = o;
}

// Transpose + convert weights.  z in [0,3]: Wq,Wk,Wv -> Wqkvt rows [z*2048, ...);
// z==3: Wo -> Wot.  Wt[n][k] = W[k][n].
__global__ __launch_bounds__(256) void wt_kernel(const float* __restrict__ Wq,
                                                 const float* __restrict__ Wk,
                                                 const float* __restrict__ Wv,
                                                 const float* __restrict__ Wo,
                                                 __bf16* __restrict__ Wqkvt,
                                                 __bf16* __restrict__ Wot) {
    __shared__ float tile[32][33];
    int z = blockIdx.z;
    const float* W = (z == 0) ? Wq : (z == 1) ? Wk : (z == 2) ? Wv : Wo;
    __bf16* Wt = (z < 3) ? (Wqkvt + (size_t)z * HIDN * HIDN) : Wot;
    int k0 = blockIdx.x * 32, n0 = blockIdx.y * 32;
    int tx = threadIdx.x & 31, ty = threadIdx.x >> 5;   // 32 x 8
#pragma unroll
    for (int i = 0; i < 4; ++i)
        tile[ty + i * 8][tx] = W[(size_t)(k0 + ty + i * 8) * HIDN + n0 + tx];
    __syncthreads();
#pragma unroll
    for (int i = 0; i < 4; ++i)
        Wt[(size_t)(n0 + ty + i * 8) * HIDN + k0 + tx] = (__bf16)tile[tx][ty + i * 8];
}

// T5 relative bias folded to per-head distance table: bias_tab[h][d], d = max(q-k,0).
__global__ __launch_bounds__(256) void bias_tab_kernel(const float* __restrict__ rel_bias,
                                                       float* __restrict__ bias_tab) {
    int d = blockIdx.x * 256 + threadIdx.x;   // 0..2047
    int h = blockIdx.y;                       // 0..31
    int bucket;
    if (d < 16) {
        bucket = d;
    } else {
        // 16 + log(d/16)/log(128/16) * 16, trunc, clamp to 31
        float f = __logf((float)d * (1.0f / 16.0f)) * (16.0f / __logf(8.0f));
        bucket = 16 + (int)f;
        if (bucket > 31) bucket = 31;
    }
    bias_tab[h * SQ + d] = rel_bias[bucket * NH + h];
}

// ---------------------------------------------------------------------------
// Generic bf16 GEMM core: C(128x128 block) = A(MxK,row-major) * Bt(NxK,row-major)^T
// 8 waves per block arranged 2(M) x 4(N); each wave: 64x32 tile = 4x2 wmma accs.
// ---------------------------------------------------------------------------
__device__ __forceinline__ void gemm_core(const __bf16* __restrict__ A,
                                          const __bf16* __restrict__ Bt,
                                          int m0, int n0, int K,
                                          int li, int hi, v8f acc[4][2]) {
#pragma unroll
    for (int mt = 0; mt < 4; ++mt)
#pragma unroll
        for (int nt = 0; nt < 2; ++nt)
#pragma unroll
            for (int i = 0; i < 8; ++i) acc[mt][nt][i] = 0.0f;

    for (int kk = 0; kk < K; kk += 32) {
        v16bf afrag[4];
#pragma unroll
        for (int mt = 0; mt < 4; ++mt) {
            const __bf16* ap = A + (size_t)(m0 + mt * 16 + li) * K + kk + hi * 16;
            bf16x8 lo = *(const bf16x8*)ap;            // K = kh*16 .. +7
            bf16x8 hi8 = *(const bf16x8*)(ap + 8);     // K = kh*16+8 .. +7
#pragma unroll
            for (int i = 0; i < 8; ++i) { afrag[mt][i] = lo[i]; afrag[mt][8 + i] = hi8[i]; }
        }
        v16bf bfrag[2];
#pragma unroll
        for (int nt = 0; nt < 2; ++nt) {
            const __bf16* bp = Bt + (size_t)(n0 + nt * 16 + li) * K + kk + hi * 16;
            bfrag[nt] = *(const v16bf*)bp;             // K = kh*16 .. +15
        }
#pragma unroll
        for (int mt = 0; mt < 4; ++mt)
#pragma unroll
            for (int nt = 0; nt < 2; ++nt)
                acc[mt][nt] = wmma_bf16(afrag[mt], bfrag[nt], acc[mt][nt]);
    }
}

// Fused QKV projection: [4096 x 6144] = Hb * WqkvtT; scatter to Q/K/Vt head layouts.
__global__ __launch_bounds__(256) void qkv_gemm_kernel(const __bf16* __restrict__ Hb,
                                                       const __bf16* __restrict__ Wqkvt,
                                                       __bf16* __restrict__ Qb,
                                                       __bf16* __restrict__ Kb,
                                                       __bf16* __restrict__ Vt) {
    int w = threadIdx.x >> 5, lane = threadIdx.x & 31;
    int li = lane & 15, hi = lane >> 4;
    int m0 = blockIdx.y * 128 + (w & 1) * 64;
    int n0 = blockIdx.x * 128 + (w >> 1) * 32;
    v8f acc[4][2];
    gemm_core(Hb, Wqkvt, m0, n0, HIDN, li, hi, acc);

#pragma unroll
    for (int mt = 0; mt < 4; ++mt)
#pragma unroll
        for (int nt = 0; nt < 2; ++nt) {
            int n = n0 + nt * 16 + li;
#pragma unroll
            for (int j = 0; j < 8; ++j) {
                int m = m0 + mt * 16 + hi * 8 + j;
                int b = m >> 11, s = m & (SQ - 1);
                __bf16 bv = (__bf16)acc[mt][nt][j];
                if (n < HIDN) {
                    int hh = n >> 6, dd = n & 63;
                    Qb[((size_t)(b * NH + hh) * SQ + s) * DKV + dd] = bv;
                } else if (n < 2 * HIDN) {
                    int n2 = n - HIDN; int hh = n2 >> 6, dd = n2 & 63;
                    Kb[((size_t)(b * NH + hh) * SQ + s) * DKV + dd] = bv;
                } else {
                    int n2 = n - 2 * HIDN; int hh = n2 >> 6, dd = n2 & 63;
                    Vt[((size_t)(b * NH + hh) * DKV + dd) * SQ + s] = bv;
                }
            }
        }
}

// Output projection: out[4096][2048] (f32) = Ab * WotT
__global__ __launch_bounds__(256) void out_gemm_kernel(const __bf16* __restrict__ Ab,
                                                       const __bf16* __restrict__ Wot,
                                                       float* __restrict__ out) {
    int w = threadIdx.x >> 5, lane = threadIdx.x & 31;
    int li = lane & 15, hi = lane >> 4;
    int m0 = blockIdx.y * 128 + (w & 1) * 64;
    int n0 = blockIdx.x * 128 + (w >> 1) * 32;
    v8f acc[4][2];
    gemm_core(Ab, Wot, m0, n0, HIDN, li, hi, acc);

#pragma unroll
    for (int mt = 0; mt < 4; ++mt)
#pragma unroll
        for (int nt = 0; nt < 2; ++nt) {
            int n = n0 + nt * 16 + li;
#pragma unroll
            for (int j = 0; j < 8; ++j) {
                int m = m0 + mt * 16 + hi * 8 + j;
                out[(size_t)m * HIDN + n] = acc[mt][nt][j];
            }
        }
}

// ---------------------------------------------------------------------------
// Flash attention. Grid: (S/128, B*NH). 8 waves; wave w owns q-rows [q0, q0+16).
// K (128x64) / V (64x128) tiles are shared by all 8 waves -> staged once per
// block into LDS with gfx1250 async global->LDS copies, double-buffered so the
// async engine prefetches block kb+1 while WMMAs process block kb.
// Online softmax; bias from bias_tab; NO causal mask (reference applies none).
// ---------------------------------------------------------------------------
__global__ __launch_bounds__(256) void flash_attn_kernel(const __bf16* __restrict__ Qb,
                                                         const __bf16* __restrict__ Kb,
                                                         const __bf16* __restrict__ Vt,
                                                         const float* __restrict__ bias_tab,
                                                         __bf16* __restrict__ Ab) {
    // Row padding (72 / 136 elements) => conflict-free ds_load_b128 fragment reads.
    __shared__ __align__(16) __bf16 Ktile[2][128][72];
    __shared__ __align__(16) __bf16 Vtile[2][64][136];
    __shared__ __align__(16) __bf16 Pl[8][16][136];

    int tid = threadIdx.x;
    int w = tid >> 5, lane = tid & 31;
    int li = lane & 15, hi = lane >> 4;
    int bh = blockIdx.y;            // 0..63
    int b = bh >> 5, h = bh & 31;
    int q0 = blockIdx.x * 128 + w * 16;

    const __bf16* Qh = Qb + (size_t)bh * SQ * DKV;
    const __bf16* Kh = Kb + (size_t)bh * SQ * DKV;
    const __bf16* Vh = Vt + (size_t)bh * DKV * SQ;
    const float*  bh_tab = bias_tab + h * SQ;

    // Per-thread staging assignment: 64 bytes each for K tile and V tile.
    int krow = tid >> 1, khalf = tid & 1;   // K tile: 128 rows x 128B
    int vrow = tid >> 2, vq = tid & 3;      // V tile: 64 rows x 256B

    // Q fragments: 2 chunks over d (K-dim 0..31, 32..63), held in registers
    v16bf qfrag[2];
#pragma unroll
    for (int c = 0; c < 2; ++c) {
        const __bf16* qp = Qh + (size_t)(q0 + li) * DKV + c * 32 + hi * 16;
        bf16x8 lo = *(const bf16x8*)qp;
        bf16x8 hi8 = *(const bf16x8*)(qp + 8);
#pragma unroll
        for (int i = 0; i < 8; ++i) { qfrag[c][i] = lo[i]; qfrag[c][8 + i] = hi8[i]; }
    }

    float m_run[8], l_run[8];
    v8f oacc[4];
#pragma unroll
    for (int j = 0; j < 8; ++j) { m_run[j] = -1e30f; l_run[j] = 0.0f; }
#pragma unroll
    for (int dt = 0; dt < 4; ++dt)
#pragma unroll
        for (int i = 0; i < 8; ++i) oacc[dt][i] = 0.0f;

    // Stage first K/V block into buffer 0.
    {
        const __bf16* gk = Kh + (size_t)krow * DKV + khalf * 32;
        const __bf16* gv = Vh + (size_t)vrow * SQ + vq * 32;
        stage_cp64(gk, &Ktile[0][krow][khalf * 32]);
        stage_cp64(gv, &Vtile[0][vrow][vq * 32]);
    }

    for (int kb = 0; kb < SQ / 128; ++kb) {
        int k0 = kb * 128;
        int buf = kb & 1;

        stage_wait();          // this wave's async copies done
        __syncthreads();       // (also flushes DScnt) -> staged tiles visible; prev reads retired

        // Prefetch next block into the other buffer while we compute.
        if (kb + 1 < SQ / 128) {
            int k1 = (kb + 1) * 128;
            const __bf16* gk = Kh + (size_t)(k1 + krow) * DKV + khalf * 32;
            const __bf16* gv = Vh + (size_t)vrow * SQ + k1 + vq * 32;
            stage_cp64(gk, &Ktile[buf ^ 1][krow][khalf * 32]);
            stage_cp64(gv, &Vtile[buf ^ 1][vrow][vq * 32]);
        }

        // --- scores: 8 tiles of 16x16, K-dim = 64 (2 wmma each), K from LDS ---
        v8f sacc[8];
#pragma unroll
        for (int t = 0; t < 8; ++t)
#pragma unroll
            for (int i = 0; i < 8; ++i) sacc[t][i] = 0.0f;
#pragma unroll
        for (int t = 0; t < 8; ++t) {
#pragma unroll
            for (int c = 0; c < 2; ++c) {
                const __bf16* kp = &Ktile[buf][t * 16 + li][c * 32 + hi * 16];
                v16bf kf = *(const v16bf*)kp;
                sacc[t] = wmma_bf16(qfrag[c], kf, sacc[t]);
            }
        }

        // --- add relative bias, compute per-row max ---
        float mx[8];
#pragma unroll
        for (int j = 0; j < 8; ++j) mx[j] = -1e30f;
#pragma unroll
        for (int t = 0; t < 8; ++t) {
            int kcol = k0 + t * 16 + li;
#pragma unroll
            for (int j = 0; j < 8; ++j) {
                int qrow = q0 + hi * 8 + j;
                int delta = qrow - kcol; if (delta < 0) delta = 0;
                float sv = sacc[t][j] + bh_tab[delta];
                sacc[t][j] = sv;
                mx[j] = fmaxf(mx[j], sv);
            }
        }
#pragma unroll
        for (int j = 0; j < 8; ++j)
#pragma unroll
            for (int off = 1; off < 16; off <<= 1)
                mx[j] = fmaxf(mx[j], __shfl_xor(mx[j], off, 32));

        // --- online softmax rescale ---
        float psum[8];
#pragma unroll
        for (int j = 0; j < 8; ++j) {
            float mnew = fmaxf(m_run[j], mx[j]);
            float alpha = __expf(m_run[j] - mnew);
            m_run[j] = mnew;
            l_run[j] *= alpha;
#pragma unroll
            for (int dt = 0; dt < 4; ++dt) oacc[dt][j] *= alpha;
            psum[j] = 0.0f;
        }

        // --- exponentiate + stash P into LDS (C/D layout -> [m][k]) ---
#pragma unroll
        for (int t = 0; t < 8; ++t)
#pragma unroll
            for (int j = 0; j < 8; ++j) {
                float p = __expf(sacc[t][j] - m_run[j]);
                psum[j] += p;
                Pl[w][hi * 8 + j][t * 16 + li] = (__bf16)p;
            }
#pragma unroll
        for (int j = 0; j < 8; ++j) {
#pragma unroll
            for (int off = 1; off < 16; off <<= 1)
                psum[j] += __shfl_xor(psum[j], off, 32);
            l_run[j] += psum[j];
        }
        __syncthreads();   // P stores visible for A-fragment reads

        // --- P @ V : A-fragments from LDS P, B-fragments from LDS V tile ---
        v16bf pfrag[4];
#pragma unroll
        for (int kc = 0; kc < 4; ++kc) {
            const __bf16* pp = &Pl[w][li][kc * 32 + hi * 16];
            bf16x8 lo = *(const bf16x8*)pp;
            bf16x8 hi8 = *(const bf16x8*)(pp + 8);
#pragma unroll
            for (int i = 0; i < 8; ++i) { pfrag[kc][i] = lo[i]; pfrag[kc][8 + i] = hi8[i]; }
        }
#pragma unroll
        for (int dt = 0; dt < 4; ++dt)
#pragma unroll
            for (int kc = 0; kc < 4; ++kc) {
                const __bf16* vp = &Vtile[buf][dt * 16 + li][kc * 32 + hi * 16];
                v16bf vf = *(const v16bf*)vp;
                oacc[dt] = wmma_bf16(pfrag[kc], vf, oacc[dt]);
            }
        // No trailing barrier needed: next iteration's top __syncthreads orders
        // these LDS reads against the buffer overwrite two iterations out.
    }

    // --- normalize and store (bf16, [b*S + s][h*64 + d] layout for out-proj) ---
#pragma unroll
    for (int dt = 0; dt < 4; ++dt)
#pragma unroll
        for (int j = 0; j < 8; ++j) {
            int s = q0 + hi * 8 + j;
            float v = oacc[dt][j] / l_run[j];
            Ab[(size_t)(b * SQ + s) * HIDN + h * DKV + dt * 16 + li] = (__bf16)v;
        }
}

// ---------------------------------------------------------------------------
// Launch
// ---------------------------------------------------------------------------
extern "C" void kernel_launch(void* const* d_in, const int* in_sizes, int n_in,
                              void* d_out, int out_size, void* d_ws, size_t ws_size,
                              hipStream_t stream) {
    const float* H  = (const float*)d_in[0];
    const float* Wq = (const float*)d_in[1];
    const float* Wk = (const float*)d_in[2];
    const float* Wv = (const float*)d_in[3];
    const float* Wo = (const float*)d_in[4];
    const float* rb = (const float*)d_in[5];
    float* out = (float*)d_out;

    char* ws = (char*)d_ws;
    __bf16* Hb    = (__bf16*)(ws + HB_OFF);
    __bf16* Wqkvt = (__bf16*)(ws + WQKVT_OFF);
    __bf16* Wot   = (__bf16*)(ws + WOT_OFF);
    __bf16* Qb    = (__bf16*)(ws + QB_OFF);
    __bf16* Kb    = (__bf16*)(ws + KB_OFF);
    __bf16* Vt    = (__bf16*)(ws + VT_OFF);
    __bf16* Ab    = (__bf16*)(ws + AB_OFF);
    float*  bias  = (float*)(ws + BIAS_OFF);

    cvt_h_kernel<<<(MROWS * HIDN) / (256 * 4), 256, 0, stream>>>(H, Hb);
    wt_kernel<<<dim3(HIDN / 32, HIDN / 32, 4), 256, 0, stream>>>(Wq, Wk, Wv, Wo, Wqkvt, Wot);
    bias_tab_kernel<<<dim3(SQ / 256, NH), 256, 0, stream>>>(rb, bias);

    qkv_gemm_kernel<<<dim3(NQKV / 128, MROWS / 128), 256, 0, stream>>>(Hb, Wqkvt, Qb, Kb, Vt);
    flash_attn_kernel<<<dim3(SQ / 128, BB * NH), 256, 0, stream>>>(Qb, Kb, Vt, bias, Ab);
    out_gemm_kernel<<<dim3(HIDN / 128, MROWS / 128), 256, 0, stream>>>(Ab, Wot, out);
}
